// SelfAttention3_23708219474452
// MI455X (gfx1250) — matrix operand order repaired and checked by reading, not compiled, not run on hardware
//
#include <hip/hip_runtime.h>
#include <hip/hip_bf16.h>
#include <math.h>

typedef __attribute__((ext_vector_type(16))) _Float16 v16h;
typedef __attribute__((ext_vector_type(8)))  float    v8f;

#define B_  32
#define S_  512
#define D_  512
#define H_  8
#define L_  4
#define AD_ 32
#define DQ_ 4    // AD/H
#define DV_ 64   // D/H

static __device__ __forceinline__ int laneId() { return threadIdx.x & 31; }
static __device__ __forceinline__ int waveId() { return threadIdx.x >> 5; }

// pack two f32 -> packed f16x2 dword
static __device__ __forceinline__ unsigned pack2(float a, float b) {
  union { _Float16 h[2]; unsigned u; } u;
  u.h[0] = (_Float16)a; u.h[1] = (_Float16)b;
  return u.u;
}

// CDNA5 async global->LDS copy, 16 bytes per lane (ASYNCcnt tracked).
// LDS address = low 32 bits of the generic pointer (flat-aperture rule).
static __device__ __forceinline__ void async_copy_b128(void* lds, const void* g) {
  unsigned l = (unsigned)(uintptr_t)lds;
  asm volatile("global_load_async_to_lds_b128 %0, %1, off"
               :: "v"(l), "v"(g) : "memory");
}
static __device__ __forceinline__ void wait_async0() {
  asm volatile("s_wait_asynccnt 0x0" ::: "memory");
}

// ---------------------------------------------------------------------------
// 1) h[b,s,:] = emb_table[x[b,s],:] + posenc(s,:)
// ---------------------------------------------------------------------------
__global__ void embed_kernel(const int* __restrict__ x,
                             const float* __restrict__ emb,
                             float* __restrict__ h) {
  const int bs = blockIdx.x;            // 0..B*S-1
  const int s  = bs % S_;
  const int row = x[bs];
  const float* e = emb + (long)row * D_;
  float* out = h + (long)bs * D_;
  const float negln1e4_over_D = -9.210340371976184f / (float)D_;
  for (int d = threadIdx.x; d < D_; d += blockDim.x) {
    int de = d & ~1;
    float div = __expf(negln1e4_over_D * (float)de);
    float ang = (float)s * div;
    float pe  = (d & 1) ? __cosf(ang) : __sinf(ang);
    out[d] = e[d] + pe;
  }
}

// ---------------------------------------------------------------------------
// 2) q = relu(h @ Wq + bq), k = relu(h @ Wk + bk).  M=16384, N=32(+32), K=512
//    WMMA f16, fp32 accumulate. Block = 8 waves, each wave owns 16 rows.
// ---------------------------------------------------------------------------
__global__ void qk_kernel(const float* __restrict__ h,
                          const float* __restrict__ Wq, const float* __restrict__ bq,
                          const float* __restrict__ Wk, const float* __restrict__ bk,
                          float* __restrict__ qbuf, float* __restrict__ kbuf) {
  __shared__ _Float16 sA[8][16][34];    // per-wave A tile, row-major [m][k], padded
  __shared__ _Float16 sWq[32][34];      // transposed weights [n][k], padded
  __shared__ _Float16 sWk[32][34];

  const int w = waveId(), L = laneId();
  const int m0 = blockIdx.x * 128 + w * 16;
  const int mrow = L & 15, half = L >> 4;

  v8f cq0 = {}, cq1 = {}, ck0 = {}, ck1 = {};

  for (int k0 = 0; k0 < D_; k0 += 32) {
    __syncthreads();
    // stage A: lane loads row (L>>1), 16 cols; packed f16x2 LDS stores
    {
      int r  = L >> 1;
      int c0 = (L & 1) * 16;
      const float* src = h + (long)(m0 + r) * D_ + k0 + c0;
      if (k0 + 32 < D_) __builtin_prefetch(src + 32, 0, 3);
      #pragma unroll
      for (int i = 0; i < 16; i += 4) {
        float4 f = *(const float4*)(src + i);
        *(unsigned*)&sA[w][r][c0 + i + 0] = pack2(f.x, f.y);
        *(unsigned*)&sA[w][r][c0 + i + 2] = pack2(f.z, f.w);
      }
    }
    // stage weight slices [k0..k0+31, 0..31], transposed; each thread owns a
    // K-pair so the packed dword is exactly the B-fragment K-pair.
    {
      int t  = threadIdx.x;
      int k2 = (t & 15) * 2;            // 0..30
      int n2 = (t >> 4) * 2;            // 0..30
      float2 qa = *(const float2*)(Wq + (long)(k0 + k2) * AD_ + n2);
      float2 qb = *(const float2*)(Wq + (long)(k0 + k2 + 1) * AD_ + n2);
      *(unsigned*)&sWq[n2 + 0][k2] = pack2(qa.x, qb.x);
      *(unsigned*)&sWq[n2 + 1][k2] = pack2(qa.y, qb.y);
      float2 ka = *(const float2*)(Wk + (long)(k0 + k2) * AD_ + n2);
      float2 kb = *(const float2*)(Wk + (long)(k0 + k2 + 1) * AD_ + n2);
      *(unsigned*)&sWk[n2 + 0][k2] = pack2(ka.x, kb.x);
      *(unsigned*)&sWk[n2 + 1][k2] = pack2(ka.y, kb.y);
    }
    __syncthreads();

    // A fragment (16x32 f16 layout: lane=M, K pairs split across lane halves)
    v16h a;
    #pragma unroll
    for (int v = 0; v < 8; v++) {
      int kb = ((v >> 2) * 16) + half * 8 + (v & 3) * 2;
      a[2 * v + 0] = sA[w][mrow][kb + 0];
      a[2 * v + 1] = sA[w][mrow][kb + 1];
    }
    // B fragments (32x16 f16: lanes 0-15 K=0-7,16-23, 2 K per VGPR)
    v16h bq0f, bq1f, bk0f, bk1f;
    #pragma unroll
    for (int v = 0; v < 8; v++) {
      int kk = ((v >> 2) * 16) + half * 8 + (v & 3) * 2;
      bq0f[2 * v] = sWq[mrow][kk];      bq0f[2 * v + 1] = sWq[mrow][kk + 1];
      bq1f[2 * v] = sWq[16 + mrow][kk]; bq1f[2 * v + 1] = sWq[16 + mrow][kk + 1];
      bk0f[2 * v] = sWk[mrow][kk];      bk0f[2 * v + 1] = sWk[mrow][kk + 1];
      bk1f[2 * v] = sWk[16 + mrow][kk]; bk1f[2 * v + 1] = sWk[16 + mrow][kk + 1];
    }
    cq0 = __builtin_amdgcn_wmma_f32_16x16x32_f16(false, a, false, bq0f, (short)0, cq0, false, false);
    cq1 = __builtin_amdgcn_wmma_f32_16x16x32_f16(false, a, false, bq1f, (short)0, cq1, false, false);
    ck0 = __builtin_amdgcn_wmma_f32_16x16x32_f16(false, a, false, bk0f, (short)0, ck0, false, false);
    ck1 = __builtin_amdgcn_wmma_f32_16x16x32_f16(false, a, false, bk1f, (short)0, ck1, false, false);
  }

  // bias + relu + store (C layout: VGPR r -> M = r + 8*half, N = lane&15)
  #pragma unroll
  for (int r = 0; r < 8; r++) {
    int row = m0 + r + half * 8;
    int n   = mrow;
    float* qo = qbuf + (long)row * AD_;
    float* ko = kbuf + (long)row * AD_;
    qo[n]      = fmaxf(cq0[r] + bq[n],      0.0f);
    qo[16 + n] = fmaxf(cq1[r] + bq[16 + n], 0.0f);
    ko[n]      = fmaxf(ck0[r] + bk[n],      0.0f);
    ko[16 + n] = fmaxf(ck1[r] + bk[16 + n], 0.0f);
  }
}

// ---------------------------------------------------------------------------
// 3) transposed-softmax statistics: for each (b,h,k) column, online max & sum
//    over all 512 queries of exp(score).  K=4 dot products in VALU.
// ---------------------------------------------------------------------------
__global__ void stats_kernel(const float* __restrict__ qbuf,
                             const float* __restrict__ kbuf,
                             float* __restrict__ colmax,
                             float* __restrict__ colsum) {
  const int bh = blockIdx.x;        // B*H
  const int b = bh >> 3, hh = bh & 7;
  __shared__ float4 sQ[S_];
  const int t = threadIdx.x;        // 512 threads = one key column each

  // async global->LDS copy of this thread's query vector (16B)
  const float* qr = qbuf + (long)(b * S_ + t) * AD_ + hh * DQ_;
  async_copy_b128(&sQ[t], qr);

  const float* kr = kbuf + (long)(b * S_ + t) * AD_ + hh * DQ_;
  float k0 = kr[0], k1 = kr[1], k2 = kr[2], k3 = kr[3];
  wait_async0();
  __syncthreads();

  const float invscale = 0.70710678118654752f;   // 1 / dq^0.25 = 1/sqrt(2)
  float m = -1e30f, sum = 0.0f;
  for (int q = 0; q < S_; q++) {
    float4 qv = sQ[q];
    float s = (qv.x * k0 + qv.y * k1 + qv.z * k2 + qv.w * k3) * invscale;
    float nm = fmaxf(m, s);
    sum = sum * __expf(m - nm) + __expf(s - nm);
    m = nm;
  }
  colmax[(long)bh * S_ + t] = m;
  colsum[(long)bh * S_ + t] = sum;
}

// ---------------------------------------------------------------------------
// 4) out[k,d] = sum_q P[q,k] * V[q,d]  (K=512 GEMM per (b,h)) via WMMA.
//    A = P^T tiles rebuilt on the fly (dot4 -> exp -> scale), V staged in LDS.
//    Block = one (b,h,128-row k-strip); 8 waves, each owns 16 k-rows x 64 d.
// ---------------------------------------------------------------------------
__global__ void av_kernel(const float* __restrict__ h,
                          const float* __restrict__ qbuf,
                          const float* __restrict__ kbuf,
                          const float* __restrict__ colmax,
                          const float* __restrict__ colsum,
                          float* __restrict__ z) {
  __shared__ float4    sQ[S_];        // all 512 query vectors (dq=4)
  __shared__ _Float16  sV[64][34];    // V chunk transposed: [d][q], padded

  const int blk = blockIdx.x;         // B*H*4
  const int b  = blk >> 5;
  const int hh = (blk >> 2) & 7;
  const int ks = (blk & 3) * 128;
  const int w = waveId(), L = laneId();
  const int t = threadIdx.x;          // 256
  const int half = L >> 4, mrow = L & 15;

  // async-stage all 512 query vectors (2 x 16B per thread)
  for (int q = t; q < S_; q += 256) {
    const float* qr = qbuf + (long)(b * S_ + q) * AD_ + hh * DQ_;
    async_copy_b128(&sQ[q], qr);
  }

  const int krow = ks + w * 16 + mrow;             // this lane's output k-row
  const float* kr = kbuf + (long)(b * S_ + krow) * AD_ + hh * DQ_;
  const float kh0 = kr[0], kh1 = kr[1], kh2 = kr[2], kh3 = kr[3];
  const float cmax = colmax[(long)(b * H_ + hh) * S_ + krow];
  const float cinv = 1.0f / colsum[(long)(b * H_ + hh) * S_ + krow];
  const float invscale = 0.70710678118654752f;
  wait_async0();

  v8f cc[4] = {{}, {}, {}, {}};

  for (int qc = 0; qc < S_; qc += 32) {
    __syncthreads();
    // stage V chunk transposed: sV[d][q] = h[b, qc+q, hh*64+d].
    // Thread owns a q-PAIR x 4 d's so each LDS store is a packed K-pair dword.
    {
      int q2 = (t >> 4) * 2;           // 0..30
      int d4 = (t & 15) * 4;           // 0..60
      const float* s0 = h + (long)(b * S_ + qc + q2) * D_ + hh * DV_ + d4;
      const float* s1 = s0 + D_;
      if (qc + 32 < S_) __builtin_prefetch(s0 + 32 * D_, 0, 3);
      float4 f0 = *(const float4*)s0;
      float4 f1 = *(const float4*)s1;
      *(unsigned*)&sV[d4 + 0][q2] = pack2(f0.x, f1.x);
      *(unsigned*)&sV[d4 + 1][q2] = pack2(f0.y, f1.y);
      *(unsigned*)&sV[d4 + 2][q2] = pack2(f0.z, f1.z);
      *(unsigned*)&sV[d4 + 3][q2] = pack2(f0.w, f1.w);
    }
    __syncthreads();

    // A = P^T tile: element (mrow, q), P = exp(s - colmax)/colsum
    v16h a;
    #pragma unroll
    for (int v = 0; v < 8; v++) {
      int kb = ((v >> 2) * 16) + half * 8 + (v & 3) * 2;
      #pragma unroll
      for (int j = 0; j < 2; j++) {
        float4 qv = sQ[qc + kb + j];
        float s = (qv.x * kh0 + qv.y * kh1 + qv.z * kh2 + qv.w * kh3) * invscale;
        a[2 * v + j] = (_Float16)(__expf(s - cmax) * cinv);
      }
    }
    // 4 B tiles over d (n = lane&15 -> d = dt*16 + n)
    #pragma unroll
    for (int dt = 0; dt < 4; dt++) {
      v16h bf;
      #pragma unroll
      for (int v = 0; v < 8; v++) {
        int kk = ((v >> 2) * 16) + half * 8 + (v & 3) * 2;
        bf[2 * v]     = sV[dt * 16 + mrow][kk];
        bf[2 * v + 1] = sV[dt * 16 + mrow][kk + 1];
      }
      cc[dt] = __builtin_amdgcn_wmma_f32_16x16x32_f16(false, a, false, bf, (short)0, cc[dt], false, false);
    }
  }

  #pragma unroll
  for (int r = 0; r < 8; r++) {
    int row = ks + w * 16 + r + half * 8;
    float* zo = z + (long)(b * S_ + row) * D_ + hh * DV_;
    zo[0 * 16 + mrow] = cc[0][r];
    zo[1 * 16 + mrow] = cc[1][r];
    zo[2 * 16 + mrow] = cc[2][r];
    zo[3 * 16 + mrow] = cc[3][r];
  }
}

// ---------------------------------------------------------------------------
// 5) h += a*(z-mean)/(std_unbiased+eps) + b    (one block per row)
// ---------------------------------------------------------------------------
__global__ void lnres_kernel(const float* __restrict__ z,
                             const float* __restrict__ ga,
                             const float* __restrict__ gb,
                             float* __restrict__ h) {
  const int row = blockIdx.x;           // B*S
  const float* zr = z + (long)row * D_;
  float* hr = h + (long)row * D_;
  const int t = threadIdx.x;            // 256
  float x0 = zr[t], x1 = zr[t + 256];

  __shared__ float red[8];
  __shared__ float sMean, sInv;

  float s = x0 + x1;
  #pragma unroll
  for (int off = 16; off > 0; off >>= 1) s += __shfl_down(s, off, 32);
  if (laneId() == 0) red[waveId()] = s;
  __syncthreads();
  if (t == 0) {
    float tot = 0.f;
    #pragma unroll
    for (int i = 0; i < 8; i++) tot += red[i];
    sMean = tot * (1.0f / 512.0f);
  }
  __syncthreads();
  float mean = sMean;
  float d0 = x0 - mean, d1 = x1 - mean;
  float v = d0 * d0 + d1 * d1;
  #pragma unroll
  for (int off = 16; off > 0; off >>= 1) v += __shfl_down(v, off, 32);
  if (laneId() == 0) red[waveId()] = v;
  __syncthreads();
  if (t == 0) {
    float tot = 0.f;
    #pragma unroll
    for (int i = 0; i < 8; i++) tot += red[i];
    float var = tot * (1.0f / 511.0f);          // unbiased (ddof=1)
    sInv = 1.0f / (sqrtf(var) + 1e-6f);
  }
  __syncthreads();
  float inv = sInv;
  hr[t]       += ga[t]       * d0 * inv + gb[t];
  hr[t + 256] += ga[t + 256] * d1 * inv + gb[t + 256];
}

// ---------------------------------------------------------------------------
// 6) pooled[b,d] = max_s h[b,s,d]; trailing scalar 1.0f
// ---------------------------------------------------------------------------
__global__ void pool_kernel(const float* __restrict__ h, float* __restrict__ out) {
  const int idx = blockIdx.x * blockDim.x + threadIdx.x;   // B*D
  const int b = idx / D_, d = idx % D_;
  float m = -INFINITY;
  for (int s = 0; s < S_; s++) m = fmaxf(m, h[(long)(b * S_ + s) * D_ + d]);
  out[idx] = m;
  if (idx == 0) out[B_ * D_] = 1.0f;
}

// ---------------------------------------------------------------------------
extern "C" void kernel_launch(void* const* d_in, const int* in_sizes, int n_in,
                              void* d_out, int out_size, void* d_ws, size_t ws_size,
                              hipStream_t stream) {
  const int*   x    = (const int*)d_in[1];
  // d_in[2] = mask (all ones -> no-op in this graph)
  const float* emb  = (const float*)d_in[3];
  const float* Wq   = (const float*)d_in[4];
  const float* bq   = (const float*)d_in[5];
  const float* Wk   = (const float*)d_in[6];
  const float* bk   = (const float*)d_in[7];
  const float* ln_a = (const float*)d_in[8];
  const float* ln_b = (const float*)d_in[9];
  float* out = (float*)d_out;

  float* ws   = (float*)d_ws;
  float* hbuf = ws;                                   // B*S*D
  float* zbuf = hbuf + (size_t)B_ * S_ * D_;          // B*S*D
  float* qbuf = zbuf + (size_t)B_ * S_ * D_;          // B*S*AD
  float* kbuf = qbuf + (size_t)B_ * S_ * AD_;         // B*S*AD
  float* cmax = kbuf + (size_t)B_ * S_ * AD_;         // B*H*S
  float* csum = cmax + (size_t)B_ * H_ * S_;          // B*H*S

  embed_kernel<<<B_ * S_, 256, 0, stream>>>(x, emb, hbuf);

  for (int l = 0; l < L_; l++) {
    qk_kernel<<<(B_ * S_) / 128, 256, 0, stream>>>(
        hbuf, Wq + (size_t)l * D_ * AD_, bq + (size_t)l * AD_,
        Wk + (size_t)l * D_ * AD_, bk + (size_t)l * AD_, qbuf, kbuf);
    stats_kernel<<<B_ * H_, 512, 0, stream>>>(qbuf, kbuf, cmax, csum);
    av_kernel<<<B_ * H_ * 4, 256, 0, stream>>>(hbuf, qbuf, kbuf, cmax, csum, zbuf);
    lnres_kernel<<<B_ * S_, 256, 0, stream>>>(
        zbuf, ln_a + (size_t)l * D_, ln_b + (size_t)l * D_, hbuf);
  }

  pool_kernel<<<(B_ * D_) / 256, 256, 0, stream>>>(hbuf, out);
}
